// LocalSelfAttention3D_18485539242793
// MI455X (gfx1250) — matrix operand order repaired and checked by reading, not compiled, not run on hardware
//
#include <hip/hip_runtime.h>

typedef __attribute__((ext_vector_type(16))) _Float16 v16h;
typedef __attribute__((ext_vector_type(8)))  _Float16 v8h;
typedef __attribute__((ext_vector_type(4)))  _Float16 v4h;
typedef __attribute__((ext_vector_type(8)))  float    v8f;

#define CDIM   256
#define NSP    32768            // 32*32*32 spatial positions
#define HD     64               // head span
#define ATTN_SCALE 0.125f       // 64^-0.5
#define XLD    264              // f16 row stride for Xs/Os/Qs/Ks (33*16B rows)
#define VLD    72               // f16 row stride for Vt (9*16B rows)
#define PLD    72               // f16 row stride for per-wave P slab
#define YLD    264              // f32 row stride for Ys

// LDS layout (bytes):
//   [0      , 33792)  Xs  (64 x 264 f16)   -- reused as Os after QKV
//   [33792  , 67584)  Qs  (64 x 264 f16)  \__ reused as Ys (64 x 264 f32)
//   [67584  , 101376) Ks  (64 x 264 f16)  /
//   [101376 , 138240) Vts (256 x 72 f16)   V stored transposed [chan][tok]
//   [138240 , 156672) P slabs: 8 waves x (16 x 72 f16)
#define SMEM_BYTES 156672

__device__ __forceinline__ v16h frag_a(const _Float16* base, int ld, int row0,
                                       int k0, int lane) {
  // A 16x32 f16: lane L holds row M = L&15; K = {kb..kb+7, kb+16..kb+23},
  // kb = (L&16) ? 8 : 0  (per ISA 16-bit A-matrix table)
  const int row = row0 + (lane & 15);
  const int kb  = k0 + ((lane & 16) >> 1);
  const _Float16* p = base + row * ld + kb;
  union { v16h v; v8h h[2]; } u;
  u.h[0] = *(const v8h*)(p);
  u.h[1] = *(const v8h*)(p + 16);
  return u.v;
}

__device__ __forceinline__ v16h frag_bt(const _Float16* base, int ld, int col0,
                                        int k0, int lane) {
  // B 32x16 f16 from a row-major [N][K] (i.e. B-transposed) source:
  // lane L holds column N = L&15, K = {kb..kb+7, kb+16..kb+23}, kb = (L&16)?8:0
  const int col = col0 + (lane & 15);
  const int kb  = k0 + ((lane & 16) >> 1);
  const _Float16* p = base + col * ld + kb;
  union { v16h v; v8h h[2]; } u;
  u.h[0] = *(const v8h*)(p);
  u.h[1] = *(const v8h*)(p + 16);
  return u.v;
}

__device__ __forceinline__ v8f wmma16(v16h a, v16h b, v8f c) {
  return __builtin_amdgcn_wmma_f32_16x16x32_f16(false, a, false, b,
                                                (short)0, c, false, false);
}

// ---- pass 0: convert weights f32 -> f16 into workspace ----
__global__ void cvt_weights_kernel(const float* __restrict__ wqkv,
                                   const float* __restrict__ wproj,
                                   _Float16* __restrict__ wh) {
  const int i = blockIdx.x * blockDim.x + threadIdx.x;
  if (i < 196608)       wh[i] = (_Float16)wqkv[i];
  else if (i < 262144)  wh[i] = (_Float16)wproj[i - 196608];
}

// ---- main: one block per (batch, window) ----
__global__ __launch_bounds__(256, 1)
void win_attn_kernel(const float* __restrict__ x,
                     const _Float16* __restrict__ wq,   // [768][256] f16
                     const _Float16* __restrict__ wp,   // [256][256] f16
                     float* __restrict__ out) {
  __shared__ __align__(16) char smem[SMEM_BYTES];
  _Float16* Xs  = (_Float16*)(smem);            // also Os
  _Float16* Qs  = (_Float16*)(smem + 33792);
  _Float16* Ks  = (_Float16*)(smem + 67584);
  float*    Ys  = (float*)   (smem + 33792);    // union over Qs+Ks
  _Float16* Vts = (_Float16*)(smem + 101376);
  _Float16* Pb  = (_Float16*)(smem + 138240);

  const int tid  = threadIdx.x;
  const int lane = tid & 31;
  // force wave id scalar: all tile routing / epilogue branches become s_cbranch
  const int wu   = __builtin_amdgcn_readfirstlane(tid >> 5);
  const int rb   = (lane & 16) >> 1;   // 0 or 8: C/D row base for this half-wave
  const int ln   = lane & 15;          // C/D column within tile

  const int blk = blockIdx.x;
  const int b   = blk >> 9;
  const int wi  = (blk >> 6) & 7;
  const int wj  = (blk >> 3) & 7;
  const int wk  = blk & 7;
  const size_t sbase = (size_t)(wi * 4) * 1024 + (size_t)(wj * 4) * 32 + (wk * 4);
  const float* xb = x   + (size_t)b * CDIM * NSP;
  float*       ob = out + (size_t)b * CDIM * NSP;

  // ---- Stage 1: gather scrambled window matrix into LDS ----
  // Xs[m][cc] = x[chan = 4m + (cc>>6)][tok = cc & 63]; 4 consecutive cc = 4
  // consecutive w positions -> float4 gather, contiguous 4xf16 store.
  for (int idx = tid; idx < 4096; idx += 256) {
    const int m  = idx >> 6;
    const int s6 = idx & 63;
    const int g  = s6 >> 4;          // cc>>6
    const int a  = (s6 >> 2) & 3;    // local d
    const int bw = s6 & 3;           // local h
    const float4 v = *(const float4*)(xb + (size_t)(4 * m + g) * NSP + sbase
                                      + a * 1024 + bw * 32);
    v4h h;
    h.x = (_Float16)v.x; h.y = (_Float16)v.y;
    h.z = (_Float16)v.z; h.w = (_Float16)v.w;
    *(v4h*)(&Xs[m * XLD + g * 64 + a * 16 + bw * 4]) = h;
  }
  __syncthreads();

  // ---- Stage 2: QKV GEMM: [64x256] @ [256x768]^T ----
  // Each wave owns 6 N-tiles; B-frags (8x v16h = 64 VGPRs) loaded once and
  // reused across the 4 M-tiles -> 4x fewer global B loads.
  for (int ntb = 0; ntb < 6; ++ntb) {
    const int n0 = (wu * 6 + ntb) << 4;
    v16h bfr[8];
#pragma unroll
    for (int ks = 0; ks < 8; ++ks)
      bfr[ks] = frag_bt(wq, 256, n0, ks * 32, lane);
#pragma unroll
    for (int mt = 0; mt < 4; ++mt) {
      const int m0 = mt << 4;
      v8f acc = {};
#pragma unroll
      for (int ks = 0; ks < 8; ++ks) {
        const v16h afr = frag_a(Xs, XLD, m0, ks * 32, lane);
        acc = wmma16(afr, bfr[ks], acc);
      }
      const int gc = n0 + ln;
      if (n0 < 256) {               // Q, row-major [tok][chan]
#pragma unroll
        for (int r = 0; r < 8; ++r)
          Qs[(m0 + rb + r) * XLD + gc] = (_Float16)acc[r];
      } else if (n0 < 512) {        // K, row-major [tok][chan]
#pragma unroll
        for (int r = 0; r < 8; ++r)
          Ks[(m0 + rb + r) * XLD + (gc - 256)] = (_Float16)acc[r];
      } else {                      // V stored transposed [chan][tok]
#pragma unroll
        for (int r = 0; r < 8; ++r)
          Vts[(gc - 512) * VLD + (m0 + rb + r)] = (_Float16)acc[r];
      }
    }
  }
  __syncthreads();

  // ---- Stage 3: attention. 16 row-blocks (head, mtile), 2 per wave. ----
  _Float16* Pw = Pb + wu * 16 * PLD;
  for (int rr = 0; rr < 2; ++rr) {
    const int rbk = wu + rr * 8;
    const int h   = rbk >> 2;
    const int m0  = (rbk & 3) << 4;
    // S = Q Kt (16x64 row block), kept in registers; Q-frags hoisted
    v16h qf[2];
#pragma unroll
    for (int kt = 0; kt < 2; ++kt)
      qf[kt] = frag_a(Qs + h * HD, XLD, m0, kt * 32, lane);
    float sv[4][8];
#pragma unroll
    for (int nt = 0; nt < 4; ++nt) {
      v8f acc = {};
#pragma unroll
      for (int kt = 0; kt < 2; ++kt) {
        const v16h kf = frag_bt(Ks + h * HD, XLD, nt * 16, kt * 32, lane);
        acc = wmma16(qf[kt], kf, acc);
      }
#pragma unroll
      for (int r = 0; r < 8; ++r) {
        float s = acc[r] * ATTN_SCALE;
        sv[nt][r] = fminf(10.0f, fmaxf(-10.0f, s));
      }
    }
    // softmax per row: reduce over 4 regs then across the 16-lane group
#pragma unroll
    for (int r = 0; r < 8; ++r) {
      float mx = fmaxf(fmaxf(sv[0][r], sv[1][r]), fmaxf(sv[2][r], sv[3][r]));
      mx = fmaxf(mx, __shfl_xor(mx, 1));
      mx = fmaxf(mx, __shfl_xor(mx, 2));
      mx = fmaxf(mx, __shfl_xor(mx, 4));
      mx = fmaxf(mx, __shfl_xor(mx, 8));
      const float e0 = __expf(sv[0][r] - mx);
      const float e1 = __expf(sv[1][r] - mx);
      const float e2 = __expf(sv[2][r] - mx);
      const float e3 = __expf(sv[3][r] - mx);
      float s = e0 + e1 + e2 + e3;
      s += __shfl_xor(s, 1);
      s += __shfl_xor(s, 2);
      s += __shfl_xor(s, 4);
      s += __shfl_xor(s, 8);
      const float inv = 1.0f / s;
      const int pr = (rb + r) * PLD + ln;
      Pw[pr +  0] = (_Float16)(e0 * inv);
      Pw[pr + 16] = (_Float16)(e1 * inv);
      Pw[pr + 32] = (_Float16)(e2 * inv);
      Pw[pr + 48] = (_Float16)(e3 * inv);
    }
    // O = P @ V_h ; P-frags hoisted; scatter with head-concat scramble into
    // Os (= Xs buffer): O_concat[h*16 + (n>>2)][(n&3)*64 + d]
    v16h pf[2];
#pragma unroll
    for (int kt = 0; kt < 2; ++kt)
      pf[kt] = frag_a(Pw, PLD, 0, kt * 32, lane);
#pragma unroll
    for (int nt = 0; nt < 4; ++nt) {
      v8f acc = {};
#pragma unroll
      for (int kt = 0; kt < 2; ++kt) {
        const v16h vf = frag_bt(Vts + h * HD * VLD, VLD, nt * 16, kt * 32, lane);
        acc = wmma16(pf[kt], vf, acc);
      }
      const int d = nt * 16 + ln;
#pragma unroll
      for (int r = 0; r < 8; ++r) {
        const int n = m0 + rb + r;
        Xs[(h * 16 + (n >> 2)) * XLD + ((n & 3) * 64 + d)] = (_Float16)acc[r];
      }
    }
  }
  __syncthreads();

  // ---- Stage 4: proj GEMM: [64x256] @ [256x256]^T -> Ys f32 ----
  // Each wave owns 2 N-tiles; B-frags loaded once, reused for 4 M-tiles.
  for (int ntb = 0; ntb < 2; ++ntb) {
    const int n0 = (wu * 2 + ntb) << 4;
    v16h bfr[8];
#pragma unroll
    for (int ks = 0; ks < 8; ++ks)
      bfr[ks] = frag_bt(wp, 256, n0, ks * 32, lane);
#pragma unroll
    for (int mt = 0; mt < 4; ++mt) {
      const int m0 = mt << 4;
      v8f acc = {};
#pragma unroll
      for (int ks = 0; ks < 8; ++ks) {
        const v16h afr = frag_a(Xs, XLD, m0, ks * 32, lane);
        acc = wmma16(afr, bfr[ks], acc);
      }
      const int gc = n0 + ln;
#pragma unroll
      for (int r = 0; r < 8; ++r)
        Ys[(m0 + rb + r) * YLD + gc] = acc[r];
    }
  }
  __syncthreads();

  // ---- Stage 5: coalesced scatter (rows are tokens, cols are channels) ----
  for (int idx = tid; idx < 4096; idx += 256) {
    const int cc  = idx >> 4;
    const int seg = idx & 15;
    const int a   = seg >> 2;
    const int bw  = seg & 3;
    const int r0  = a * 16 + bw * 4;
    float4 v;
    v.x = Ys[(r0 + 0) * YLD + cc];
    v.y = Ys[(r0 + 1) * YLD + cc];
    v.z = Ys[(r0 + 2) * YLD + cc];
    v.w = Ys[(r0 + 3) * YLD + cc];
    *(float4*)(ob + (size_t)cc * NSP + sbase + a * 1024 + bw * 32) = v;
  }
}

extern "C" void kernel_launch(void* const* d_in, const int* in_sizes, int n_in,
                              void* d_out, int out_size, void* d_ws,
                              size_t ws_size, hipStream_t stream) {
  (void)in_sizes; (void)n_in; (void)out_size; (void)ws_size;
  const float* x     = (const float*)d_in[0];
  const float* wqkv  = (const float*)d_in[1];
  const float* wproj = (const float*)d_in[2];
  float* out         = (float*)d_out;
  _Float16* wh       = (_Float16*)d_ws;   // 262144 f16 = 512 KB

  cvt_weights_kernel<<<1024, 256, 0, stream>>>(wqkv, wproj, wh);
  win_attn_kernel<<<2048, 256, 0, stream>>>(x, wh, wh + 196608, out);
}
